// GEN_Composer_7705171329661
// MI455X (gfx1250) — matrix operand order repaired and checked by reading, not compiled, not run on hardware
//
#include <hip/hip_runtime.h>
#include <hip/hip_bf16.h>

#define BB 4
#define NN 10000
#define EE 160000
#define QQ 256
#define DD 128
#define MM 64
#define UU 126
#define HH 128
#define PP 7
#define OUTC (PP + DD) /* 135 */
#define NKX 313        /* ceil(10000/32) k-steps for readout GEMM */

typedef __attribute__((ext_vector_type(16))) __bf16 v16bf;
typedef __attribute__((ext_vector_type(8)))  __bf16 v8bf;
typedef __attribute__((ext_vector_type(8)))  float  v8f;

__device__ __forceinline__ v8f wmma_bf16(v16bf a, v16bf b, v8f c) {
  return __builtin_amdgcn_wmma_f32_16x16x32_bf16(false, a, false, b, (short)0, c,
                                                 false, false);
}

__device__ __forceinline__ v16bf cat8(v8bf lo, v8bf hi) {
  v16bf r;
#pragma unroll
  for (int i = 0; i < 8; i++) { r[i] = lo[i]; r[i + 8] = hi[i]; }
  return r;
}

// convert 16 contiguous f32 -> 16 bf16, store to 16B-aligned dst (2x ds_store_b128)
__device__ __forceinline__ void cvt16_store(const float* __restrict__ src,
                                            __bf16* dst) {
  const float4* p = (const float4*)src;
  float4 a = p[0], b = p[1], c = p[2], d = p[3];
  float f[16] = {a.x, a.y, a.z, a.w, b.x, b.y, b.z, b.w,
                 c.x, c.y, c.z, c.w, d.x, d.y, d.z, d.w};
  v8bf v0, v1;
#pragma unroll
  for (int i = 0; i < 8; i++) {
    v0[i] = (__bf16)f[i];
    v1[i] = (__bf16)f[i + 8];
  }
  *(v8bf*)dst = v0;
  *(v8bf*)(dst + 8) = v1;
}

// ---------------- prep kernels ----------------
__global__ void k_copy_f32(const float* __restrict__ in, float* __restrict__ out, int n) {
  int i = blockIdx.x * blockDim.x + threadIdx.x;
  if (i < n) out[i] = in[i];
}
__global__ void k_zero_f32(float* __restrict__ p, int n) {
  int i = blockIdx.x * blockDim.x + threadIdx.x;
  if (i < n) p[i] = 0.f;
}

// Pack weight W[K][Ncols] (f32 row-major) into WMMA B-fragment order, bf16.
// out[((nTile*nK + kt)*32 + lane)*16 + i] = W[kt*32 + 8*(lane>>4) + (i<8 ? i : i+8)][nTile*16 + (lane&15)]
__global__ void k_pack_w(const float* __restrict__ w, __bf16* __restrict__ out,
                         int K, int Ncols, int NcolsPad) {
  int idx = blockIdx.x * blockDim.x + threadIdx.x;
  int nK = K / 32;
  int total = (NcolsPad / 16) * nK * 512;
  if (idx >= total) return;
  int i     = idx & 15;
  int lane  = (idx >> 4) & 31;
  int frag  = idx >> 9;
  int kt    = frag % nK;
  int nTile = frag / nK;
  int k   = kt * 32 + ((lane >> 4) * 8) + ((i < 8) ? i : (i + 8));
  int col = nTile * 16 + (lane & 15);
  out[idx] = (col < Ncols) ? (__bf16)w[(size_t)k * Ncols + col] : (__bf16)0.f;
}

// Pack attn[B,Q,N] into WMMA A-fragment order (per q-tile), zero-padded K to 10016.
// A layout: lane row=(lane&15); elems 0..7 -> K = kt*32+8*hi+i ; 8..15 -> +16
__global__ void k_pack_attn(const float* __restrict__ attn, __bf16* __restrict__ out) {
  int idx = blockIdx.x * blockDim.x + threadIdx.x;
  int total = BB * 16 * NKX * 512;
  if (idx >= total) return;
  int i    = idx & 15;
  int lane = (idx >> 4) & 31;
  int frag = idx >> 9;
  int kt   = frag % NKX;
  int qt   = (frag / NKX) & 15;
  int b    = frag / (NKX * 16);
  int hi   = lane >> 4;
  int k    = kt * 32 + 8 * hi + ((i < 8) ? i : (i + 8));
  int q    = qt * 16 + (lane & 15);
  out[idx] = (k < NN) ? (__bf16)attn[((size_t)b * QQ + q) * NN + k] : (__bf16)0.f;
}

// Pack final states[B,N,D] into WMMA B-fragment order for the readout GEMM.
__global__ void k_pack_states(const float* __restrict__ states, __bf16* __restrict__ out) {
  int idx = blockIdx.x * blockDim.x + threadIdx.x;
  int total = BB * 8 * NKX * 512;
  if (idx >= total) return;
  int i    = idx & 15;
  int lane = (idx >> 4) & 31;
  int frag = idx >> 9;
  int kt   = frag % NKX;
  int ct   = (frag / NKX) & 7;
  int b    = frag / (NKX * 8);
  int k    = kt * 32 + 8 * (lane >> 4) + ((i < 8) ? i : (i + 8));
  int col  = ct * 16 + (lane & 15);
  out[idx] = (k < NN) ? (__bf16)states[((size_t)b * NN + k) * DD + col] : (__bf16)0.f;
}

// ---------------- edge MLP + scatter ----------------
// 1 block = 16 edges of one batch; 8 waves; einp staged once in LDS as bf16.
__global__ __launch_bounds__(256) void k_edge_mlp(
    const float* __restrict__ states,
    const int* __restrict__ esrc, const int* __restrict__ esnk,
    const __bf16* __restrict__ w1p, const float* __restrict__ b1e,
    const __bf16* __restrict__ w2p, const float* __restrict__ b2e,
    float* __restrict__ incoming) {
  int tile = blockIdx.x;  // B * (E/16)
  int b    = tile / (EE / 16);
  int e0   = (tile % (EE / 16)) * 16;
  int t    = threadIdx.x;
  int lane = t & 31;
  int wave = t >> 5;
  int row  = lane & 15;
  int hi   = lane >> 4;

  __shared__ __align__(16) __bf16 einp[16][272];  // 16 edges x 256, padded
  __shared__ __align__(16) __bf16 act[16][144];   // 16 edges x 128, padded
  __shared__ int sink16[16];

  // stage einp: thread t -> edge t>>4, 16-value segment t&15
  {
    int erow = t >> 4;
    int seg  = t & 15;
    int e    = e0 + erow;
    int node = (seg < 8) ? esrc[e] : esnk[e];
    if (t < 16) sink16[t] = esnk[e0 + t];
    const float* src = states + ((size_t)b * NN + node) * DD + (seg & 7) * 16;
    cvt16_store(src, &einp[erow][seg * 16]);
  }
  __syncthreads();

  // layer 1: [16x256] @ [256x128]; wave owns H cols [16*wave, 16*wave+16)
  v8f acc = {};
  {
    const v16bf* wp = (const v16bf*)w1p + ((size_t)wave * 8) * 32 + lane;
    for (int kt = 0; kt < 8; kt++) {
      int ka   = kt * 32 + 8 * hi;
      v8bf lo  = *(const v8bf*)&einp[row][ka];
      v8bf hh  = *(const v8bf*)&einp[row][ka + 16];
      acc = wmma_bf16(cat8(lo, hh), wp[kt * 32], acc);
    }
  }
  {
    float bias = b1e[wave * 16 + row];
#pragma unroll
    for (int r = 0; r < 8; r++) {  // C: row r(+8*hi), col = lane&15
      float v = acc[r] + bias;
      act[r + 8 * hi][wave * 16 + row] = (__bf16)(v > 0.f ? v : 0.f);
    }
  }
  __syncthreads();

  // layer 2: [16x128] @ [128x64]; waves 0..3 (wave-uniform -> EXEC all-1s)
  if (wave < 4) {
    v8f acc2 = {};
    const v16bf* wp = (const v16bf*)w2p + ((size_t)wave * 4) * 32 + lane;
    for (int kt = 0; kt < 4; kt++) {
      int ka  = kt * 32 + 8 * hi;
      v8bf lo = *(const v8bf*)&act[row][ka];
      v8bf hh = *(const v8bf*)&act[row][ka + 16];
      acc2 = wmma_bf16(cat8(lo, hh), wp[kt * 32], acc2);
    }
    int col    = wave * 16 + row;
    float bias = b2e[col];
#pragma unroll
    for (int r = 0; r < 8; r++) {
      int erow = r + 8 * hi;
      atomicAdd(incoming + ((size_t)b * NN + sink16[erow]) * MM + col, acc2[r] + bias);
    }
  }
}

// ---------------- node MLP + in-place state update ----------------
__global__ __launch_bounds__(256) void k_node_mlp(
    float* __restrict__ states, const float* __restrict__ incoming,
    const __bf16* __restrict__ w1p, const float* __restrict__ b1n,
    const __bf16* __restrict__ w2p, const float* __restrict__ b2n) {
  int tile = blockIdx.x;  // B * (N/16)
  int b    = tile / (NN / 16);
  int n0   = (tile % (NN / 16)) * 16;
  int t    = threadIdx.x;
  int lane = t & 31;
  int wave = t >> 5;
  int row  = lane & 15;
  int hi   = lane >> 4;

  __shared__ __align__(16) __bf16 ninp[16][208];  // 16 nodes x 192, padded
  __shared__ __align__(16) __bf16 act[16][144];

  // stage ninp = [incoming(64) | states(128)]: 192 threads, 16-value segments
  if (t < 192) {  // waves 0..5 fully active (wave-uniform)
    int nrow = t / 12;
    int seg  = t % 12;
    const float* src =
        (seg < 4) ? incoming + ((size_t)b * NN + n0 + nrow) * MM + seg * 16
                  : states + ((size_t)b * NN + n0 + nrow) * DD + (seg - 4) * 16;
    cvt16_store(src, &ninp[nrow][seg * 16]);
  }
  __syncthreads();

  // layer 1: [16x192] @ [192x128]
  v8f acc = {};
  {
    const v16bf* wp = (const v16bf*)w1p + ((size_t)wave * 6) * 32 + lane;
    for (int kt = 0; kt < 6; kt++) {
      int ka  = kt * 32 + 8 * hi;
      v8bf lo = *(const v8bf*)&ninp[row][ka];
      v8bf hh = *(const v8bf*)&ninp[row][ka + 16];
      acc = wmma_bf16(cat8(lo, hh), wp[kt * 32], acc);
    }
  }
  {
    float bias = b1n[wave * 16 + row];
#pragma unroll
    for (int r = 0; r < 8; r++) {
      float v = acc[r] + bias;
      act[r + 8 * hi][wave * 16 + row] = (__bf16)(v > 0.f ? v : 0.f);
    }
  }
  __syncthreads();

  // layer 2: [16x128] @ [128x128 (padded from 126)]
  v8f acc2 = {};
  {
    const v16bf* wp = (const v16bf*)w2p + ((size_t)wave * 4) * 32 + lane;
    for (int kt = 0; kt < 4; kt++) {
      int ka  = kt * 32 + 8 * hi;
      v8bf lo = *(const v8bf*)&act[row][ka];
      v8bf hh = *(const v8bf*)&act[row][ka + 16];
      acc2 = wmma_bf16(cat8(lo, hh), wp[kt * 32], acc2);
    }
  }
  int col = wave * 16 + row;
  if (col < UU) {  // valid cols 0..125 -> channels 2..127
    float bias = b2n[col];
#pragma unroll
    for (int r = 0; r < 8; r++) {
      int nrow   = r + 8 * hi;
      float* dst = states + ((size_t)b * NN + n0 + nrow) * DD + (DD - UU) + col;
      *dst += acc2[r] + bias;
    }
  }
}

// ---------------- readout: out = [poses | attn @ states] ----------------
// All operands pre-packed into fragment order; inner loop = 2x v16bf load + wmma.
__global__ __launch_bounds__(256) void k_extract(
    const __bf16* __restrict__ attnPack, const __bf16* __restrict__ statePack,
    const float* __restrict__ poses, float* __restrict__ out) {
  int blk  = blockIdx.x;  // B * (Q/16)
  int b    = blk >> 4;
  int qt   = blk & 15;
  int q0   = qt * 16;
  int lane = threadIdx.x & 31;
  int wave = threadIdx.x >> 5;  // column tile of D=128
  int row  = lane & 15;
  int hi   = lane >> 4;

  const v16bf* aP = (const v16bf*)attnPack + ((size_t)(b * 16 + qt) * NKX) * 32 + lane;
  const v16bf* bP = (const v16bf*)statePack + ((size_t)(b * 8 + wave) * NKX) * 32 + lane;

  v8f acc = {};
  for (int kt = 0; kt < NKX; kt++) {
    acc = wmma_bf16(aP[(size_t)kt * 32], bP[(size_t)kt * 32], acc);
  }
  int col = wave * 16 + row;
#pragma unroll
  for (int r = 0; r < 8; r++) {
    int q = q0 + r + 8 * hi;
    out[((size_t)b * QQ + q) * OUTC + PP + col] = acc[r];
  }
  if (threadIdx.x < 16 * PP) {
    int q = q0 + threadIdx.x / PP, p = threadIdx.x % PP;
    out[((size_t)b * QQ + q) * OUTC + p] = poses[((size_t)b * QQ + q) * PP + p];
  }
}

// ---------------- launcher ----------------
extern "C" void kernel_launch(void* const* d_in, const int* in_sizes, int n_in,
                              void* d_out, int out_size, void* d_ws, size_t ws_size,
                              hipStream_t stream) {
  const float* nodes = (const float*)d_in[0];
  const float* poses = (const float*)d_in[1];
  const float* attn  = (const float*)d_in[2];
  const int*   esrc  = (const int*)d_in[3];
  const int*   esnk  = (const int*)d_in[4];
  const float* w1e   = (const float*)d_in[5];
  const float* b1e   = (const float*)d_in[6];
  const float* w2e   = (const float*)d_in[7];
  const float* b2e   = (const float*)d_in[8];
  const float* w1n   = (const float*)d_in[9];
  const float* b1n   = (const float*)d_in[10];
  const float* w2n   = (const float*)d_in[11];
  const float* b2n   = (const float*)d_in[12];

  char* ws   = (char*)d_ws;
  size_t off = 0;
  auto alloc = [&](size_t bytes) {
    void* p = ws + off;
    off = (off + bytes + 255) & ~(size_t)255;
    return p;
  };
  float*  states   = (float*)alloc((size_t)BB * NN * DD * 4);
  float*  incoming = (float*)alloc((size_t)BB * NN * MM * 4);
  __bf16* w1e_b    = (__bf16*)alloc((size_t)2 * DD * HH * 2);
  __bf16* w2e_b    = (__bf16*)alloc((size_t)HH * MM * 2);
  __bf16* w1n_b    = (__bf16*)alloc((size_t)(DD + MM) * HH * 2);
  __bf16* w2n_b    = (__bf16*)alloc((size_t)HH * HH * 2);
  __bf16* attnP    = (__bf16*)alloc((size_t)BB * 16 * NKX * 512 * 2);
  __bf16* stateP   = (__bf16*)alloc((size_t)BB * 8 * NKX * 512 * 2);

  int nStates = BB * NN * DD;
  k_copy_f32<<<(nStates + 255) / 256, 256, 0, stream>>>(nodes, states, nStates);
  k_pack_w<<<(2 * DD * HH + 255) / 256, 256, 0, stream>>>(w1e, w1e_b, 2 * DD, HH, HH);
  k_pack_w<<<(HH * MM + 255) / 256, 256, 0, stream>>>(w2e, w2e_b, HH, MM, MM);
  k_pack_w<<<((DD + MM) * HH + 255) / 256, 256, 0, stream>>>(w1n, w1n_b, DD + MM, HH, HH);
  k_pack_w<<<(HH * HH + 255) / 256, 256, 0, stream>>>(w2n, w2n_b, HH, UU, HH);
  {
    int nA = BB * 16 * NKX * 512;
    k_pack_attn<<<(nA + 255) / 256, 256, 0, stream>>>(attn, attnP);
  }

  int nInc = BB * NN * MM;
  for (int s = 0; s < 3; s++) {
    k_zero_f32<<<(nInc + 255) / 256, 256, 0, stream>>>(incoming, nInc);
    k_edge_mlp<<<BB * (EE / 16), 256, 0, stream>>>(states, esrc, esnk, w1e_b, b1e,
                                                   w2e_b, b2e, incoming);
    k_node_mlp<<<BB * (NN / 16), 256, 0, stream>>>(states, incoming, w1n_b, b1n,
                                                   w2n_b, b2n);
  }
  {
    int nS = BB * 8 * NKX * 512;
    k_pack_states<<<(nS + 255) / 256, 256, 0, stream>>>(states, stateP);
  }
  k_extract<<<BB * (QQ / 16), 256, 0, stream>>>(attnP, stateP, poses, (float*)d_out);

  (void)in_sizes; (void)n_in; (void)out_size; (void)ws_size;
}
